// EncoderLayer_2370821947934
// MI455X (gfx1250) — matrix operand order repaired and checked by reading, not compiled, not run on hardware
//
#include <hip/hip_runtime.h>
#include <hip/hip_bf16.h>
#include <math.h>

// MI455X (gfx1250) CDNA5 transformer encoder layer.
// Compute-bound (~172 GFLOP vs ~9us of HBM traffic at 23.3 TB/s) -> all GEMMs
// run on v_wmma_f32_16x16x32_bf16 (bf16 in, fp32 accumulate). Softmax / LN /
// RoPE / epilogues in fp32. Tile staging: Tensor Data Mover DMA
// (tensor_load_to_lds / TENSORcnt) double-buffered in the main GEMM so the
// DMA overlaps the WMMA pipe; async global->LDS (ASYNCcnt) double-buffered in
// the GEGLU GEMM; manual load+ds_store fallback. Workspace: ~160 MB.

#define DEVI __device__ __forceinline__

constexpr int Bc  = 2;
constexpr int Sc  = 2048;
constexpr int Dc  = 1024;
constexpr int Hc  = 16;
constexpr int FFc = 4096;
constexpr int DHc = 64;
constexpr int Mrows = Bc * Sc;       // 4096 token rows
constexpr float EPSc = 1e-5f;

typedef __attribute__((ext_vector_type(16))) __bf16 v16bf;
typedef __attribute__((ext_vector_type(8)))  float  v8f;

#if defined(__AMDGCN__) && __has_builtin(__builtin_amdgcn_tensor_load_to_lds) && \
    __has_builtin(__builtin_amdgcn_s_wait_tensorcnt)
#define USE_TDM 1
#endif
#if defined(__AMDGCN__) && __has_builtin(__builtin_amdgcn_global_load_async_to_lds_b128)
#define USE_ASYNC 1
#endif

union FragBF {
  v16bf v;
  uint4 q[2];
};

DEVI v8f vzero() { v8f z; for (int i = 0; i < 8; ++i) z[i] = 0.f; return z; }

DEVI unsigned short f2bf(float f) {
  union { float f; unsigned u; } x; x.f = f;
  unsigned u = x.u;
  unsigned r = u + 0x7FFFu + ((u >> 16) & 1u);   // round-to-nearest-even
  return (unsigned short)(r >> 16);
}

DEVI v8f wmma_bf16(const FragBF& a, const FragBF& b, v8f c) {
  // (neg_a, A, neg_b, B, c_mod, C, reuse_a, reuse_b)
  return __builtin_amdgcn_wmma_f32_16x16x32_bf16(
      false, a.v, false, b.v, (short)0, c, false, false);
}

#ifdef USE_TDM
typedef unsigned int tdm_v4u __attribute__((vector_size(16)));
typedef int          tdm_v8i __attribute__((vector_size(32)));
typedef int          tdm_v4i __attribute__((vector_size(16)));

// 2D bf16 tile DMA: global (tileH rows x tileW elems, row stride strideElems)
// -> LDS at byte offset ldsOff (rows stored contiguously). D# packing per
// CDNA5 ISA ch.8: group0 {count=1, lds_addr, global_addr, type=2},
// group1 {data_size=1(2B), tensor_dim0/1, tile_dim0/1, tensor_dim0_stride}.
DEVI void tdm_load_tile2d(unsigned ldsOff, const void* g,
                          unsigned tileW, unsigned tileH,
                          unsigned tensorW, unsigned tensorH,
                          unsigned strideElems)
{
  unsigned long long ga = (unsigned long long)(size_t)g;
  tdm_v4u g0;
  g0[0] = 1u;                                        // count=1, user mode
  g0[1] = ldsOff;                                    // lds_addr (bytes)
  g0[2] = (unsigned)ga;                              // global_addr[31:0]
  g0[3] = (unsigned)((ga >> 32) & 0x01FFFFFFu) | (2u << 30);  // addr[56:32]|type=2
  tdm_v8i g1;
  g1[0] = (int)(1u << 16);                           // data_size=1 -> 2 bytes
  g1[1] = (int)((tensorW & 0xFFFFu) << 16);          // tensor_dim0[15:0] @bit48
  g1[2] = (int)((tensorW >> 16) | ((tensorH & 0xFFFFu) << 16));
  g1[3] = (int)((tensorH >> 16) | ((tileW & 0xFFFFu) << 16));   // tile_dim0
  g1[4] = (int)(tileH & 0xFFFFu);                    // tile_dim1 (tile_dim2=0)
  g1[5] = (int)strideElems;                          // tensor_dim0_stride[31:0]
  g1[6] = 0;                                         // stride[47:32], dim1_stride lo
  g1[7] = 0;
  tdm_v4i z4 = {0, 0, 0, 0};
  tdm_v8i z8 = {0, 0, 0, 0, 0, 0, 0, 0};
  __builtin_amdgcn_tensor_load_to_lds(g0, g1, z4, z4, z8, 0);
}
#define LDS_OFF(p) ((unsigned)(size_t)(__attribute__((address_space(3))) void*)(p))
#endif

#ifdef USE_ASYNC
typedef int v4i_vs __attribute__((vector_size(16)));
DEVI void async_b128(const void* g, void* lds) {
  __builtin_amdgcn_global_load_async_to_lds_b128(
      (__attribute__((address_space(1))) v4i_vs*)(g),
      (__attribute__((address_space(3))) v4i_vs*)(lds), 0, 0);
}
template <int N>
DEVI void wait_async() {
#if __has_builtin(__builtin_amdgcn_s_wait_asynccnt)
  __builtin_amdgcn_s_wait_asynccnt(N);               // N is a constant expr
#else
  asm volatile("s_wait_asynccnt %0" :: "i"(N) : "memory");
#endif
}
#endif

// ---------------------------------------------------------------- convert
__global__ void cvt_bf16_kernel(const float* __restrict__ src,
                                unsigned short* __restrict__ dst, int n) {
  int i = blockIdx.x * blockDim.x + threadIdx.x;
  if (i < n) dst[i] = f2bf(src[i]);
}

// ---------------------------------------------------------------- GEMM
// C[M,N] = A[M,K](bf16) * W[N,K]^T(bf16) + bias[N] (+ resid[M,N])
// Block tile 128x128, 8 waves (4x2) of 32x64 each, K-step 32. 8 WMMA/wave/step.
// TDM path is double-buffered: wave 0 prefetches tile k+1 while all waves
// run WMMAs on tile k (TENSORcnt<=2 allows the newest DMA pair in flight).
__global__ __launch_bounds__(256) void gemm_bias_kernel(
    const unsigned short* __restrict__ A,
    const unsigned short* __restrict__ W,
    const float* __restrict__ bias,
    const float* __restrict__ resid,
    float* __restrict__ outF,
    unsigned short* __restrict__ outB,
    int M, int N, int K)
{
  __shared__ __align__(16) unsigned short As[2][128][32];
  __shared__ __align__(16) unsigned short Bs[2][128][32];
  const int tid  = threadIdx.x;
  const int lane = tid & 31, wave = tid >> 5;
  const int wm = wave >> 1, wn = wave & 1;
  const int l16 = lane & 15, hi = lane >> 4;
  const int bm = blockIdx.y, bn = blockIdx.x;

  const unsigned short* Arow = A + (size_t)(bm * 128) * K;
  const unsigned short* Wrow = W + (size_t)(bn * 128) * K;

  v8f acc[2][4];
  for (int i = 0; i < 2; ++i)
    for (int j = 0; j < 4; ++j) acc[i][j] = vzero();

  int cur = 0;
#ifdef USE_TDM
  if (wave == 0) {                       // prologue DMA for k0 = 0
    tdm_load_tile2d(LDS_OFF(&As[0][0][0]), Arow, 32, 128, 32, 128, (unsigned)K);
    tdm_load_tile2d(LDS_OFF(&Bs[0][0][0]), Wrow, 32, 128, 32, 128, (unsigned)K);
  }
#endif

  for (int k0 = 0; k0 < K; k0 += 32) {
#ifdef USE_TDM
    if (wave == 0) {
      if (k0 + 32 < K) {                 // prefetch next tile into other buffer
        tdm_load_tile2d(LDS_OFF(&As[cur ^ 1][0][0]), Arow + k0 + 32,
                        32, 128, 32, 128, (unsigned)K);
        tdm_load_tile2d(LDS_OFF(&Bs[cur ^ 1][0][0]), Wrow + k0 + 32,
                        32, 128, 32, 128, (unsigned)K);
        __builtin_amdgcn_s_wait_tensorcnt(2);   // current pair done (in-order)
      } else {
        __builtin_amdgcn_s_wait_tensorcnt(0);
      }
    }
    __syncthreads();
#else
    {   // stage A+B tiles, 128x32 each; every thread moves 2x16 elems
      int r = tid >> 1, cb = (tid & 1) * 16;
      const unsigned short* ga = Arow + (size_t)r * K + k0 + cb;
      const unsigned short* gb = Wrow + (size_t)r * K + k0 + cb;
#ifdef USE_ASYNC
      async_b128(ga,     &As[0][r][cb]);
      async_b128(ga + 8, &As[0][r][cb + 8]);
      async_b128(gb,     &Bs[0][r][cb]);
      async_b128(gb + 8, &Bs[0][r][cb + 8]);
      wait_async<0>();
#else
      const uint4* ga4 = (const uint4*)ga;
      const uint4* gb4 = (const uint4*)gb;
      *(uint4*)&As[0][r][cb]     = ga4[0];
      *(uint4*)&As[0][r][cb + 8] = ga4[1];
      *(uint4*)&Bs[0][r][cb]     = gb4[0];
      *(uint4*)&Bs[0][r][cb + 8] = gb4[1];
#endif
    }
    __syncthreads();
#endif

    FragBF af[2], bfg[4];
    for (int i = 0; i < 2; ++i) {           // A layout: K = {hi*8..+8} U {+16}
      int row = wm*32 + i*16 + l16;
      int kb  = hi * 8;
      af[i].q[0] = *(const uint4*)&As[cur][row][kb];
      af[i].q[1] = *(const uint4*)&As[cur][row][kb + 16];
    }
    for (int j = 0; j < 4; ++j) {           // B layout: lane=N, K = {hi*8..+8} U {+16}
      int col = wn*64 + j*16 + l16;
      int kb  = hi * 8;
      bfg[j].q[0] = *(const uint4*)&Bs[cur][col][kb];
      bfg[j].q[1] = *(const uint4*)&Bs[cur][col][kb + 16];
    }
    for (int i = 0; i < 2; ++i)
      for (int j = 0; j < 4; ++j)
        acc[i][j] = wmma_bf16(af[i], bfg[j], acc[i][j]);
    __syncthreads();                       // all reads done before buffer reuse
#ifdef USE_TDM
    cur ^= 1;
#endif
  }

  // C layout: VGPR r, lanes 0-15 -> M=r, lanes 16-31 -> M=r+8, N=lane%16
  for (int i = 0; i < 2; ++i)
    for (int j = 0; j < 4; ++j) {
      int col = bn*128 + wn*64 + j*16 + l16;
      float bv = bias ? bias[col] : 0.f;
      for (int r = 0; r < 8; ++r) {
        int row = bm*128 + wm*32 + i*16 + r + 8*hi;
        size_t idx = (size_t)row * N + col;
        float v = acc[i][j][r] + bv;
        if (resid) v += resid[idx];
        if (outF)  outF[idx] = v;
        if (outB)  outB[idx] = f2bf(v);
      }
    }
}

// ------------------------------------------------------- GEGLU GEMM (W1)
// m[M,FF] = gelu(A*W1[0:FF]^T + b1[0:FF]) * (A*W1[FF:2FF]^T + b1[FF:2FF])
// Async global->LDS staging, double-buffered (ASYNCcnt<=4 keeps the newest
// 4 per-thread copies in flight while tile k is consumed).
__global__ __launch_bounds__(256) void geglu_gemm_kernel(
    const unsigned short* __restrict__ A,     // M x K
    const unsigned short* __restrict__ W1b,   // 2FF x K
    const float* __restrict__ b1,
    unsigned short* __restrict__ mb,          // M x FF (bf16)
    int M, int K)
{
  __shared__ __align__(16) unsigned short As[2][128][32];
  __shared__ __align__(16) unsigned short Ba[2][64][32];
  __shared__ __align__(16) unsigned short Bg[2][64][32];
  const int tid  = threadIdx.x;
  const int lane = tid & 31, wave = tid >> 5;
  const int wm = wave >> 1, wn = wave & 1;
  const int l16 = lane & 15, hi = lane >> 4;
  const int bm = blockIdx.y, bn = blockIdx.x;

  const int r  = tid >> 1, cb  = (tid & 1) * 16;
  const int r2 = tid >> 2, cb2 = (tid & 3) * 8;
  const unsigned short* Arow  = A   + (size_t)(bm*128 + r)*K + cb;
  const unsigned short* Wa    = W1b + (size_t)(bn*64 + r2)*K + cb2;
  const unsigned short* Wg    = W1b + (size_t)(FFc + bn*64 + r2)*K + cb2;

  v8f accA[2][2], accG[2][2];
  for (int i = 0; i < 2; ++i)
    for (int j = 0; j < 2; ++j) { accA[i][j] = vzero(); accG[i][j] = vzero(); }

  int cur = 0;
#ifdef USE_ASYNC
  async_b128(Arow,     &As[0][r][cb]);
  async_b128(Arow + 8, &As[0][r][cb + 8]);
  async_b128(Wa,       &Ba[0][r2][cb2]);
  async_b128(Wg,       &Bg[0][r2][cb2]);
#endif

  for (int k0 = 0; k0 < K; k0 += 32) {
#ifdef USE_ASYNC
    if (k0 + 32 < K) {                    // prefetch next tile
      async_b128(Arow + k0 + 32,     &As[cur ^ 1][r][cb]);
      async_b128(Arow + k0 + 32 + 8, &As[cur ^ 1][r][cb + 8]);
      async_b128(Wa + k0 + 32,       &Ba[cur ^ 1][r2][cb2]);
      async_b128(Wg + k0 + 32,       &Bg[cur ^ 1][r2][cb2]);
      wait_async<4>();                    // current tile's 4 copies done
    } else {
      wait_async<0>();
    }
    __syncthreads();
#else
    *(uint4*)&As[0][r][cb]     = *(const uint4*)(Arow + k0);
    *(uint4*)&As[0][r][cb + 8] = *(const uint4*)(Arow + k0 + 8);
    *(uint4*)&Ba[0][r2][cb2]   = *(const uint4*)(Wa + k0);
    *(uint4*)&Bg[0][r2][cb2]   = *(const uint4*)(Wg + k0);
    __syncthreads();
#endif

    FragBF af[2], ba[2], bg[2];
    for (int i = 0; i < 2; ++i) {
      int row = wm*32 + i*16 + l16;
      int kb  = hi * 8;
      af[i].q[0] = *(const uint4*)&As[cur][row][kb];
      af[i].q[1] = *(const uint4*)&As[cur][row][kb + 16];
    }
    for (int j = 0; j < 2; ++j) {
      int col = wn*32 + j*16 + l16;
      int kb  = hi * 8;
      ba[j].q[0] = *(const uint4*)&Ba[cur][col][kb];
      ba[j].q[1] = *(const uint4*)&Ba[cur][col][kb + 16];
      bg[j].q[0] = *(const uint4*)&Bg[cur][col][kb];
      bg[j].q[1] = *(const uint4*)&Bg[cur][col][kb + 16];
    }
    for (int i = 0; i < 2; ++i)
      for (int j = 0; j < 2; ++j) {
        accA[i][j] = wmma_bf16(af[i], ba[j], accA[i][j]);
        accG[i][j] = wmma_bf16(af[i], bg[j], accG[i][j]);
      }
    __syncthreads();
#ifdef USE_ASYNC
    cur ^= 1;
#endif
  }

  for (int i = 0; i < 2; ++i)
    for (int j = 0; j < 2; ++j) {
      int col = bn*64 + wn*32 + j*16 + l16;
      float bva = b1[col], bvg = b1[FFc + col];
      for (int rr = 0; rr < 8; ++rr) {
        int row = bm*128 + wm*32 + i*16 + rr + 8*hi;
        float a  = accA[i][j][rr] + bva;
        float g  = accG[i][j][rr] + bvg;
        float ge = 0.5f * a * (1.f + erff(a * 0.70710678118654752f));  // exact gelu
        mb[(size_t)row * FFc + col] = f2bf(ge * g);
      }
    }
}

// ---------------------------------------------------------------- RoPE
// qkv fp32 (B,S,3,H,DH) -> q,k bf16 (B,H,S,DH) with rotation, v bf16 (B,H,DH,S)
__global__ void rope_kernel(const float* __restrict__ qkv, const int* __restrict__ p,
                            const float* __restrict__ freqs,
                            unsigned short* __restrict__ qo,
                            unsigned short* __restrict__ ko,
                            unsigned short* __restrict__ vt)
{
  int t = blockIdx.x * blockDim.x + threadIdx.x;   // B*H*S*DH threads exactly
  int d = t & (DHc - 1);
  int s = (t >> 6) & (Sc - 1);
  int h = (t >> 17) & (Hc - 1);
  int b = t >> 21;
  const float* row = qkv + (size_t)(b * Sc + s) * (3 * Dc);
  float ang = (float)p[b * Sc + s] * freqs[d];
  float c = cosf(ang), sn = sinf(ang);
  int dro = (d < DHc / 2) ? d + DHc / 2 : d - DHc / 2;
  float qv = row[h * DHc + d],       qr = row[h * DHc + dro];
  float kv = row[Dc + h * DHc + d],  kr = row[Dc + h * DHc + dro];
  float vv = row[2 * Dc + h * DHc + d];
  size_t oi = ((size_t)(b * Hc + h) * Sc + s) * DHc + d;
  qo[oi] = f2bf(qv * c + qr * sn);
  ko[oi] = f2bf(kv * c + kr * sn);
  vt[((size_t)(b * Hc + h) * DHc + d) * Sc + s] = f2bf(vv);
}

// ------------------------------------------------------- flash attention
// One wave per (b, h, 16-row q tile). Streams K/V in 32-column steps with
// online softmax; P restaged via per-wave LDS into the A-fragment layout.
__global__ __launch_bounds__(256) void attn_kernel(
    const unsigned short* __restrict__ Qb,   // (B,H,S,DH) bf16
    const unsigned short* __restrict__ Kb,   // (B,H,S,DH) bf16
    const unsigned short* __restrict__ Vt,   // (B,H,DH,S) bf16
    unsigned short* __restrict__ Ob)         // (B,S,D)    bf16
{
  __shared__ __align__(16) unsigned short Pl[8][16 * 32];   // per-wave P tile
  const int tid = threadIdx.x, lane = tid & 31, wave = tid >> 5;
  const int l16 = lane & 15, hi = lane >> 4;
  const int gw  = blockIdx.x * 8 + wave;           // 0 .. B*H*(S/16)-1
  const int q16 = gw & (Sc / 16 - 1);
  const int bh  = gw >> 7;                         // S/16 == 128
  const int b = bh / Hc, h = bh % Hc;

  const unsigned short* Q  = Qb + ((size_t)bh * Sc + q16 * 16) * DHc;
  const unsigned short* Kp = Kb + (size_t)bh * Sc * DHc;
  const unsigned short* Vp = Vt + (size_t)bh * DHc * Sc;

  FragBF qf[2];                                    // Q: 16 x 64 as two K=32 frags
  for (int i = 0; i < 2; ++i) {
    const unsigned short* qr = Q + l16 * DHc + i * 32 + hi * 8;
    qf[i].q[0] = *(const uint4*)qr;
    qf[i].q[1] = *(const uint4*)(qr + 16);
  }

  float mrowv[8], lrow[8];
  v8f oacc[4];
  for (int r = 0; r < 8; ++r) { mrowv[r] = -INFINITY; lrow[r] = 0.f; }
  for (int t = 0; t < 4; ++t) oacc[t] = vzero();
  const float scale = 0.125f;                      // 1/sqrt(64)

  for (int ks = 0; ks < Sc; ks += 32) {
    v8f sc[2]; sc[0] = vzero(); sc[1] = vzero();
    for (int j = 0; j < 2; ++j)                    // two 16-wide key tiles
      for (int i = 0; i < 2; ++i) {                // two 32-deep d halves
        FragBF kf;
        const unsigned short* kr =
            Kp + (size_t)(ks + j * 16 + l16) * DHc + i * 32 + hi * 8;
        kf.q[0] = *(const uint4*)kr;
        kf.q[1] = *(const uint4*)(kr + 16);
        sc[j] = wmma_bf16(qf[i], kf, sc[j]);
      }

    for (int r = 0; r < 8; ++r) {                  // online softmax per row
      float s0 = sc[0][r] * scale;
      float s1 = sc[1][r] * scale;
      float mloc = fmaxf(s0, s1);
      for (int m = 1; m < 16; m <<= 1) mloc = fmaxf(mloc, __shfl_xor(mloc, m, 32));
      float mnew  = fmaxf(mrowv[r], mloc);
      float alpha = expf(mrowv[r] - mnew);
      float e0 = expf(s0 - mnew);
      float e1 = expf(s1 - mnew);
      float sloc = e0 + e1;
      for (int m = 1; m < 16; m <<= 1) sloc += __shfl_xor(sloc, m, 32);
      lrow[r]  = lrow[r] * alpha + sloc;
      mrowv[r] = mnew;
      for (int t = 0; t < 4; ++t) oacc[t][r] *= alpha;
      int prow = r + 8 * hi;
      Pl[wave][prow * 32 + l16]      = f2bf(e0);
      Pl[wave][prow * 32 + 16 + l16] = f2bf(e1);
    }
    __syncthreads();

    FragBF pf;                                     // P as A-fragment (16x32)
    {
      int kb = hi * 8;
      pf.q[0] = *(const uint4*)&Pl[wave][l16 * 32 + kb];
      pf.q[1] = *(const uint4*)&Pl[wave][l16 * 32 + kb + 16];
    }
    for (int t = 0; t < 4; ++t) {                  // O += P @ V (4 d-tiles)
      FragBF vf;
      const unsigned short* vr = Vp + (size_t)(t * 16 + l16) * Sc + ks + hi * 8;
      vf.q[0] = *(const uint4*)vr;
      vf.q[1] = *(const uint4*)(vr + 16);
      oacc[t] = wmma_bf16(pf, vf, oacc[t]);
    }
    __syncthreads();
  }

  for (int t = 0; t < 4; ++t)
    for (int r = 0; r < 8; ++r) {
      int srow = q16 * 16 + r + 8 * hi;
      int col  = h * DHc + t * 16 + l16;
      float v = oacc[t][r] / lrow[r];
      Ob[((size_t)b * Sc + srow) * Dc + col] = f2bf(v);
    }
}

// ---------------------------------------------------------------- LayerNorm
__global__ __launch_bounds__(256) void ln_kernel(
    const float* __restrict__ xin, const float* __restrict__ g,
    const float* __restrict__ bta, float* __restrict__ outF,
    unsigned short* __restrict__ outB)
{
  __shared__ float red[8];
  const int row = blockIdx.x;
  const int tid = threadIdx.x, lane = tid & 31, wave = tid >> 5;
  float4 v = ((const float4*)(xin + (size_t)row * Dc))[tid];   // 256*4 = D

  float s = v.x + v.y + v.z + v.w;
  for (int m = 16; m >= 1; m >>= 1) s += __shfl_xor(s, m, 32);
  if (lane == 0) red[wave] = s;
  __syncthreads();
  float mu = 0.f;
  for (int i = 0; i < 8; ++i) mu += red[i];
  mu *= (1.f / Dc);
  __syncthreads();

  float dx = v.x - mu, dy = v.y - mu, dz = v.z - mu, dw = v.w - mu;
  float s2 = dx*dx + dy*dy + dz*dz + dw*dw;
  for (int m = 16; m >= 1; m >>= 1) s2 += __shfl_xor(s2, m, 32);
  if (lane == 0) red[wave] = s2;
  __syncthreads();
  float var = 0.f;
  for (int i = 0; i < 8; ++i) var += red[i];
  var *= (1.f / Dc);
  float inv = rsqrtf(var + EPSc);

  float4 gg = ((const float4*)g)[tid];
  float4 bb = ((const float4*)bta)[tid];
  float4 o;
  o.x = dx * inv * gg.x + bb.x;
  o.y = dy * inv * gg.y + bb.y;
  o.z = dz * inv * gg.z + bb.z;
  o.w = dw * inv * gg.w + bb.w;
  if (outF) ((float4*)(outF + (size_t)row * Dc))[tid] = o;
  if (outB) {
    unsigned short* ob = outB + (size_t)row * Dc + tid * 4;
    ob[0] = f2bf(o.x); ob[1] = f2bf(o.y); ob[2] = f2bf(o.z); ob[3] = f2bf(o.w);
  }
}

// ---------------------------------------------------------------- launch
extern "C" void kernel_launch(void* const* d_in, const int* in_sizes, int n_in,
                              void* d_out, int out_size, void* d_ws, size_t ws_size,
                              hipStream_t stream)
{
  const float* x     = (const float*)d_in[0];
  const int*   p     = (const int*)  d_in[1];
  const float* Wqkv  = (const float*)d_in[2];
  const float* bqkv  = (const float*)d_in[3];
  const float* Wo    = (const float*)d_in[4];
  const float* bo    = (const float*)d_in[5];
  const float* g_at  = (const float*)d_in[6];
  const float* b_at  = (const float*)d_in[7];
  const float* W1    = (const float*)d_in[8];
  const float* b1    = (const float*)d_in[9];
  const float* W2    = (const float*)d_in[10];
  const float* b2    = (const float*)d_in[11];
  const float* g_ml  = (const float*)d_in[12];
  const float* b_ml  = (const float*)d_in[13];
  const float* freqs = (const float*)d_in[14];
  float* out = (float*)d_out;
  (void)in_sizes; (void)n_in; (void)out_size; (void)ws_size;

  char* ws = (char*)d_ws;
  size_t off = 0;
  auto take = [&](size_t bytes) {
    char* q = ws + off;
    off = (off + bytes + 255) & ~(size_t)255;
    return q;
  };
  unsigned short* xb   = (unsigned short*)take((size_t)Mrows * Dc * 2);
  unsigned short* wqb  = (unsigned short*)take((size_t)3 * Dc * Dc * 2);
  unsigned short* wob  = (unsigned short*)take((size_t)Dc * Dc * 2);
  unsigned short* w1b  = (unsigned short*)take((size_t)2 * FFc * Dc * 2);
  unsigned short* w2b  = (unsigned short*)take((size_t)Dc * FFc * 2);
  float*          qkvf = (float*)         take((size_t)Mrows * 3 * Dc * 4);
  unsigned short* qbuf = (unsigned short*)take((size_t)Mrows * Dc * 2);
  unsigned short* kbuf = (unsigned short*)take((size_t)Mrows * Dc * 2);
  unsigned short* vtb  = (unsigned short*)take((size_t)Mrows * Dc * 2);
  unsigned short* obuf = (unsigned short*)take((size_t)Mrows * Dc * 2);
  float*          res  = (float*)         take((size_t)Mrows * Dc * 4);
  float*          hbuf = (float*)         take((size_t)Mrows * Dc * 4);
  unsigned short* hb   = (unsigned short*)take((size_t)Mrows * Dc * 2);
  unsigned short* mb   = (unsigned short*)qkvf;   // reuse dead qkv region (32MB < 48MB)

  auto cvt = [&](const float* s, unsigned short* d, int n) {
    cvt_bf16_kernel<<<(n + 255) / 256, 256, 0, stream>>>(s, d, n);
  };
  cvt(x,    xb,  Mrows * Dc);
  cvt(Wqkv, wqb, 3 * Dc * Dc);
  cvt(Wo,   wob, Dc * Dc);
  cvt(W1,   w1b, 2 * FFc * Dc);
  cvt(W2,   w2b, Dc * FFc);

  // qkv = x @ Wqkv^T + bqkv  (fp32 out, consumed by RoPE)
  gemm_bias_kernel<<<dim3(3 * Dc / 128, Mrows / 128), 256, 0, stream>>>(
      xb, wqb, bqkv, nullptr, qkvf, nullptr, Mrows, 3 * Dc, Dc);

  rope_kernel<<<(Bc * Hc * Sc * DHc) / 256, 256, 0, stream>>>(
      qkvf, p, freqs, qbuf, kbuf, vtb);

  attn_kernel<<<(Bc * Hc * (Sc / 16)) / 8, 256, 0, stream>>>(
      qbuf, kbuf, vtb, obuf);

  // res = o @ Wo^T + bo + x
  gemm_bias_kernel<<<dim3(Dc / 128, Mrows / 128), 256, 0, stream>>>(
      obuf, wob, bo, x, res, nullptr, Mrows, Dc, Dc);

  // h = LN(res) -> fp32 + bf16
  ln_kernel<<<Mrows, 256, 0, stream>>>(res, g_at, b_at, hbuf, hb);

  // m = gelu(h@W1a^T + b1a) * (h@W1g^T + b1g)   (bf16 out)
  geglu_gemm_kernel<<<dim3(FFc / 64, Mrows / 128), 256, 0, stream>>>(
      hb, w1b, b1, mb, Mrows, Dc);

  // res = m @ W2^T + b2 + h
  gemm_bias_kernel<<<dim3(Dc / 128, Mrows / 128), 256, 0, stream>>>(
      mb, w2b, b2, hbuf, res, nullptr, Mrows, Dc, FFc);

  // out = LN(res)
  ln_kernel<<<Mrows, 256, 0, stream>>>(res, g_ml, b_ml, out, nullptr);
}
